// Model_SAGE_15616501088835
// MI455X (gfx1250) — matrix operand order repaired and checked by reading, not compiled, not run on hardware
//
#include <hip/hip_runtime.h>
#include <hip/hip_bf16.h>
#include <math.h>

// ---------------------------------------------------------------------------
// Problem constants (from the reference)
// ---------------------------------------------------------------------------
#define NN      65536        // nodes (B*NPER)
#define EE      524288       // edges (B*EPER)
#define BB      64           // graphs
#define TT      256          // TIME_L
#define DIMD    128
#define MODE_N  8
#define TOPK    20
#define LSTM_H  64

// Tensor Data Mover usage for GEMM A-tiles. Set to 0 to fall back to manual
// staging if the TDM builtin fails to compile on this toolchain.
#ifndef ENABLE_TDM
#define ENABLE_TDM 1
#endif
#if ENABLE_TDM
// therock-10.0 headers (6-arg builtin) ship this header; ROCm 7.2 (5-arg) doesn't.
#if __has_include(<hip/amd_detail/amd_gfx1250_TDM.h>)
#define TDM_6ARG 1
#else
#define TDM_6ARG 0
#endif
#endif

// ---------------------------------------------------------------------------
// WMMA / TDM types (CDNA5 / gfx1250, wave32)
// ---------------------------------------------------------------------------
typedef __attribute__((ext_vector_type(16))) __bf16    v16bf;
typedef __attribute__((ext_vector_type(8)))  __bf16    v8bf;
typedef __attribute__((ext_vector_type(8)))  float     v8f;
typedef __attribute__((ext_vector_type(4)))  unsigned  u32x4;
typedef __attribute__((ext_vector_type(8)))  int       i32x8;
typedef __attribute__((ext_vector_type(4)))  int       i32x4;

// float -> bf16 bits, round-to-nearest-even
static __device__ __forceinline__ unsigned f2bf(float f) {
    unsigned u = __float_as_uint(f);
    return (u + 0x7FFFu + ((u >> 16) & 1u)) >> 16;
}
static __device__ __forceinline__ unsigned pk2bf(float lo, float hi) {
    return (f2bf(lo) & 0xFFFFu) | (f2bf(hi) << 16);
}

#if ENABLE_TDM
// ---------------------------------------------------------------------------
// Issue a TDM 2D tile load: tileM x 32 fp32 elements, row stride K elements,
// from global `gaddr` into LDS `lds`. D# built per CDNA5 ISA ch.8:
//  group0: count=1 | lds_addr | global_addr[56:0] | type=2
//  group1: data_size=2 (4B) | tensor_dim0=K | tensor_dim1=tileM |
//          tile_dim0=32 | tile_dim1=tileM | tensor_dim0_stride=K
// ---------------------------------------------------------------------------
static __device__ __forceinline__ void tdm_load_tile(const float* gaddr,
                                                     void* lds, int K, int tileM)
{
    unsigned long long ga = (unsigned long long)(size_t)gaddr;
    unsigned ldsOff = (unsigned)(size_t)lds;     // flat LDS addr truncates to offset
    u32x4 g0;
    g0[0] = 1u;                                              // count=1 (valid D#)
    g0[1] = ldsOff;                                          // lds_addr
    g0[2] = (unsigned)(ga & 0xffffffffull);                  // global_addr[31:0]
    g0[3] = (unsigned)((ga >> 32) & 0x01ffffffull) | (2u << 30); // [56:32]|type=2
    i32x8 g1;
    g1[0] = (int)(2u << 16);                                 // data_size=4B
    g1[1] = (int)(((unsigned)K & 0xffffu) << 16);            // tensor_dim0 lo16
    g1[2] = (int)((((unsigned)K) >> 16) | ((unsigned)tileM << 16)); // dim0 hi | dim1 lo
    g1[3] = (int)(32u << 16);                                // dim1 hi(0) | tile_dim0=32
    g1[4] = (int)(unsigned)tileM;                            // tile_dim1 | tile_dim2=0
    g1[5] = K;                                               // tensor_dim0_stride lo32
    g1[6] = 0;
    g1[7] = 0;
    i32x4 z4; z4[0] = z4[1] = z4[2] = z4[3] = 0;
#if TDM_6ARG
    i32x8 z8;
    #pragma unroll
    for (int i = 0; i < 8; ++i) z8[i] = 0;
    __builtin_amdgcn_tensor_load_to_lds(g0, g1, z4, z4, z8, 0);
#else
    __builtin_amdgcn_tensor_load_to_lds(g0, g1, z4, z4, 0);
#endif
}
#endif // ENABLE_TDM

// ---------------------------------------------------------------------------
// GEMM:  C[M,N] = act( A[M,K] @ W[K,N]  (+ A2[M,K] @ W2[K,N])  + bias )
// bf16 inputs (packed 2/dword in LDS), f32 accumulate via
// v_wmma_f32_16x16x32_bf16. Block = 256 threads = 8 waves, 128x64 tile.
// Requires M % 128 == 0 and K % 32 == 0 (true for every call site here).
// ACT: 0=none 1=relu 2=tanh.  GN: guard N (only needed when N % 64 != 0).
// ---------------------------------------------------------------------------
template<int ACT, bool GN>
__global__ __launch_bounds__(256)
void gemm_wmma(const float* __restrict__ A,  const float* __restrict__ W,
               const float* __restrict__ A2, const float* __restrict__ W2,
               const float* __restrict__ bias, float* __restrict__ C,
               int M, int N, int K, int K2)
{
#if ENABLE_TDM
    __shared__ __align__(32) float    Af[128 * 32];   // TDM fp32 A tile
#endif
    __shared__ __align__(32) unsigned Ab[128 * 16];   // [m][k/2] packed bf16
    __shared__ __align__(32) unsigned Bb[64 * 16];    // [n][k/2] packed bf16 (W^T)

    const int tid  = threadIdx.x;
    const int wave = tid >> 5;
    const int lane = tid & 31;
    const int hh   = lane >> 4;          // lane half
    const int ln   = lane & 15;
    const int mBase = blockIdx.y * 128;
    const int nBase = blockIdx.x * 64;

    v8f acc[4];
    #pragma unroll
    for (int nt = 0; nt < 4; ++nt)
        #pragma unroll
        for (int r = 0; r < 8; ++r) acc[nt][r] = 0.0f;

    for (int pass = 0; pass < 2; ++pass) {
        const float* Ap = pass ? A2 : A;
        const float* Wp = pass ? W2 : W;
        const int    Kp = pass ? K2 : K;
        if (Ap == nullptr) continue;

        for (int k0 = 0; k0 < Kp; k0 += 32) {
#if ENABLE_TDM
            // ---- async DMA of A tile (one wave issues; TENSORcnt tracked) ----
            if (tid < 32)
                tdm_load_tile(Ap + (size_t)mBase * Kp + k0, Af, Kp, 128);
#else
            // ---- manual A stage: 128x32, guard-free (M%128==0, K%32==0) ----
            #pragma unroll
            for (int it = 0; it < 4; ++it) {
                int li4 = it * 256 + tid;            // 0..1023 float4 slots
                int r   = li4 >> 3;                  // row 0..127
                int c4  = (li4 & 7) * 4;             // col 0,4..28
                float4 v = *(const float4*)(Ap + (size_t)(mBase + r) * Kp + k0 + c4);
                Ab[r * 16 + (c4 >> 1)]     = pk2bf(v.x, v.y);
                Ab[r * 16 + (c4 >> 1) + 1] = pk2bf(v.z, v.w);
            }
#endif
            // ---- W stage 32x64 -> Bb[n][k] (transposed, packed) ----
            #pragma unroll
            for (int it = 0; it < 4; ++it) {
                int li = it * 256 + tid;             // 0..1023 dword slots
                int n  = li >> 4;                    // 0..63
                int k2 = (li & 15) * 2;              // 0,2..30
                int col = nBase + n;
                float v0 = 0.f, v1 = 0.f;
                if (!GN || col < N) {
                    v0 = Wp[(size_t)(k0 + k2) * N + col];
                    v1 = Wp[(size_t)(k0 + k2 + 1) * N + col];
                }
                Bb[n * 16 + (k2 >> 1)] = pk2bf(v0, v1);
            }
#if ENABLE_TDM
            if (tid < 32) __builtin_amdgcn_s_wait_tensorcnt(0);
            __syncthreads();                         // publish TDM-written Af
            // ---- convert fp32 tile -> packed bf16 ----
            #pragma unroll
            for (int it = 0; it < 8; ++it) {
                int li = it * 256 + tid;             // 0..2047 packed slots
                float2 v = *(const float2*)&Af[li * 2];
                Ab[li] = pk2bf(v.x, v.y);
            }
#endif
            __syncthreads();

            // ---- fragments + WMMA ----
            // A (16-bit 16x32): K=e+16h  (M=ln)
            const unsigned short* Ar =
                (const unsigned short*)&Ab[(wave * 16 + ln) * 16];
            v16bf a = *(const v16bf*)(Ar + hh * 16);
            #pragma unroll
            for (int nt = 0; nt < 4; ++nt) {
                // B (16-bit 32x16): lane-half holds contiguous K run (N=ln)
                const unsigned short* Br =
                    (const unsigned short*)&Bb[(nt * 16 + ln) * 16];
                v16bf b = *(const v16bf*)(Br + hh * 16);
                acc[nt] = __builtin_amdgcn_wmma_f32_16x16x32_bf16(
                              false, a, false, b, (short)0, acc[nt],
                              false, false);
            }
            __syncthreads();
        }
    }

    // ---- epilogue: C/D layout row = r + 8h, col = ln ----
    #pragma unroll
    for (int nt = 0; nt < 4; ++nt) {
        int col = nBase + nt * 16 + ln;
        if (GN && col >= N) continue;
        float bv = bias ? bias[col] : 0.0f;
        #pragma unroll
        for (int r = 0; r < 8; ++r) {
            int row = mBase + wave * 16 + r + 8 * hh;
            float v = acc[nt][r] + bv;
            if (ACT == 1) v = fmaxf(v, 0.0f);
            else if (ACT == 2) v = tanhf(v);
            C[(size_t)row * N + col] = v;
        }
    }
}

// ---------------------------------------------------------------------------
// zero-fill
// ---------------------------------------------------------------------------
__global__ void zero_kernel(float* p, size_t n) {
    size_t i = (size_t)blockIdx.x * blockDim.x + threadIdx.x;
    if (i < n) p[i] = 0.0f;
}

// ---------------------------------------------------------------------------
// segment-max scatter:  agg[dst[e]] = max(agg[dst[e]], m[src[e]])  per feature.
// m >= 0 (relu output), agg zero-init -> uint atomicMax on float bits is exact
// (matches DGL's 0-fill for empty segments). 4 features / thread via float4.
// ---------------------------------------------------------------------------
__global__ __launch_bounds__(256)
void scatter_max(const float* __restrict__ m, const int* __restrict__ src,
                 const int* __restrict__ dst, float* __restrict__ agg,
                 int D, int E)
{
    const int  tpe   = D >> 2;                       // threads per edge
    const long gid   = (long)blockIdx.x * blockDim.x + threadIdx.x;
    const long total = (long)E * tpe;
    if (gid >= total) return;
    const int e  = (int)(gid / tpe);
    const int f4 = (int)(gid % tpe) * 4;
    const int s = src[e], d = dst[e];
    const float* mp = m + (long)s * D + f4;
    __builtin_prefetch(mp + D, 0, 1);                // gfx1250 global_prefetch_b8
    const float4 v = *(const float4*)mp;
    unsigned* ap = (unsigned*)(agg + (long)d * D + f4);
    atomicMax(ap + 0, __float_as_uint(v.x));
    atomicMax(ap + 1, __float_as_uint(v.y));
    atomicMax(ap + 2, __float_as_uint(v.z));
    atomicMax(ap + 3, __float_as_uint(v.w));
}

// ---------------------------------------------------------------------------
// per-graph top-k by iterative argmax (stable: lowest index wins ties)
// ---------------------------------------------------------------------------
__global__ __launch_bounds__(256)
void topk_kernel(const float* __restrict__ scores, int* __restrict__ idx)
{
    const int b = blockIdx.x, t = threadIdx.x;
    __shared__ float sv[1024];
    __shared__ float rmax[256];
    __shared__ int   rid[256];
    for (int i = t; i < 1024; i += 256) sv[i] = scores[b * 1024 + i];
    __syncthreads();
    for (int k = 0; k < TOPK; ++k) {
        float best = -INFINITY; int bi = 1024;
        for (int i = t; i < 1024; i += 256) {
            float v = sv[i];
            if (v > best || (v == best && i < bi)) { best = v; bi = i; }
        }
        rmax[t] = best; rid[t] = bi;
        __syncthreads();
        for (int s = 128; s > 0; s >>= 1) {
            if (t < s) {
                if (rmax[t + s] > rmax[t] ||
                    (rmax[t + s] == rmax[t] && rid[t + s] < rid[t])) {
                    rmax[t] = rmax[t + s]; rid[t] = rid[t + s];
                }
            }
            __syncthreads();
        }
        if (t == 0) { idx[b * TOPK + k] = rid[0]; sv[rid[0]] = -INFINITY; }
        __syncthreads();
    }
}

// ---------------------------------------------------------------------------
// gather selected node features -> graph_spatial [B][T][TOPK]
// ---------------------------------------------------------------------------
__global__ void gather_sel(const float* __restrict__ ns,
                           const int* __restrict__ idx, float* __restrict__ gs)
{
    int i = blockIdx.x * blockDim.x + threadIdx.x;     // B*T*TOPK
    if (i >= BB * TT * TOPK) return;
    int k = i % TOPK;
    int t = (i / TOPK) % TT;
    int b = i / (TOPK * TT);
    int node = b * 1024 + idx[b * TOPK + k];
    gs[i] = ns[(long)node * TT + t];
}

// ---------------------------------------------------------------------------
// one LSTM layer, PyTorch gate order (i,f,g,o). One block per batch element,
// 256 threads = one gate pre-activation each; h/c/z staged in LDS.
// Wih: [4H, in_sz] row-major, Whh: [4H, H].
// ---------------------------------------------------------------------------
__global__ __launch_bounds__(256)
void lstm_layer(const float* __restrict__ x,      // [B,T,in_sz]
                const float* __restrict__ Wih, const float* __restrict__ Whh,
                const float* __restrict__ bih, const float* __restrict__ bhh,
                float* __restrict__ y,            // [B,T,H]
                int in_sz, int T)
{
    const int b = blockIdx.x, j = threadIdx.x;
    __shared__ float xs[LSTM_H], hs[LSTM_H], cs[LSTM_H], zs[256];
    if (j < LSTM_H) { hs[j] = 0.f; cs[j] = 0.f; }
    __syncthreads();
    const float bsum = bih[j] + bhh[j];
    const float* wi = Wih + (long)j * in_sz;
    const float* wh = Whh + (long)j * LSTM_H;
    for (int t = 0; t < T; ++t) {
        if (j < in_sz) xs[j] = x[((long)b * T + t) * in_sz + j];
        __syncthreads();
        float z = bsum;
        for (int i = 0; i < in_sz; ++i) z += wi[i] * xs[i];
        #pragma unroll 8
        for (int i = 0; i < LSTM_H; ++i) z += wh[i] * hs[i];
        zs[j] = z;
        __syncthreads();
        if (j < LSTM_H) {
            float ig = 1.f / (1.f + __expf(-zs[j]));
            float fg = 1.f / (1.f + __expf(-zs[64 + j]));
            float g  = tanhf(zs[128 + j]);
            float og = 1.f / (1.f + __expf(-zs[192 + j]));
            float c  = fg * cs[j] + ig * g;
            float h  = og * tanhf(c);
            cs[j] = c; hs[j] = h;
            y[((long)b * T + t) * LSTM_H + j] = h;
        }
        __syncthreads();
    }
}

// ---------------------------------------------------------------------------
// phi column abs-max (uint trick on |x| bits) + scale
// ---------------------------------------------------------------------------
__global__ void col_absmax(const float* __restrict__ phi, unsigned* cm, int M) {
    int i = blockIdx.x * blockDim.x + threadIdx.x;
    if (i < M * MODE_N) atomicMax(&cm[i & (MODE_N - 1)],
                                  __float_as_uint(fabsf(phi[i])));
}
__global__ void col_scale(float* phi, const unsigned* cm, int M) {
    int i = blockIdx.x * blockDim.x + threadIdx.x;
    if (i < M * MODE_N) phi[i] /= __uint_as_float(cm[i & (MODE_N - 1)]);
}

// ---------------------------------------------------------------------------
// host orchestration
// ---------------------------------------------------------------------------
extern "C" void kernel_launch(void* const* d_in, const int* in_sizes, int n_in,
                              void* d_out, int out_size, void* d_ws, size_t ws_size,
                              hipStream_t stream)
{
    (void)in_sizes; (void)n_in; (void)out_size; (void)ws_size;

    // ---- inputs (setup_inputs() dict insertion order; nested dicts likewise)
    const float* acc_Y = (const float*)d_in[0];        // [N, 256]
    const int*   srcI  = (const int*)d_in[1];          // [E]
    const int*   dstI  = (const int*)d_in[2];          // [E]
    int p = 3;
    // sage dicts: Wp, bp, Ws, Wn, b
    const float *c1_Wp=(const float*)d_in[p++], *c1_bp=(const float*)d_in[p++],
                *c1_Ws=(const float*)d_in[p++], *c1_Wn=(const float*)d_in[p++],
                *c1_b =(const float*)d_in[p++];
    const float *c2_Wp=(const float*)d_in[p++], *c2_bp=(const float*)d_in[p++],
                *c2_Ws=(const float*)d_in[p++], *c2_Wn=(const float*)d_in[p++],
                *c2_b =(const float*)d_in[p++];
    const float *c3_Wp=(const float*)d_in[p++], *c3_bp=(const float*)d_in[p++],
                *c3_Ws=(const float*)d_in[p++], *c3_Wn=(const float*)d_in[p++],
                *c3_b =(const float*)d_in[p++];
    const float *sc_Wp=(const float*)d_in[p++], *sc_bp=(const float*)d_in[p++],
                *sc_Ws=(const float*)d_in[p++], *sc_Wn=(const float*)d_in[p++],
                *sc_b =(const float*)d_in[p++];
    // lstm layers: Wih, Whh, bih, bhh
    const float *l_Wih[4], *l_Whh[4], *l_bih[4], *l_bhh[4];
    for (int L = 0; L < 4; ++L) {
        l_Wih[L]=(const float*)d_in[p++]; l_Whh[L]=(const float*)d_in[p++];
        l_bih[L]=(const float*)d_in[p++]; l_bhh[L]=(const float*)d_in[p++];
    }
    // mlp dicts: W1,b1,W2,b2,W3,b3
    const float *de_W1=(const float*)d_in[p++], *de_b1=(const float*)d_in[p++],
                *de_W2=(const float*)d_in[p++], *de_b2=(const float*)d_in[p++],
                *de_W3=(const float*)d_in[p++], *de_b3=(const float*)d_in[p++];
    const float *nd_W1=(const float*)d_in[p++], *nd_b1=(const float*)d_in[p++],
                *nd_W2=(const float*)d_in[p++], *nd_b2=(const float*)d_in[p++],
                *nd_W3=(const float*)d_in[p++], *nd_b3=(const float*)d_in[p++];

    // ---- workspace carve (~280 MB)
    char* wsp = (char*)d_ws;
    auto carve = [&](size_t bytes) -> void* {
        void* q = (void*)wsp; wsp += (bytes + 255) & ~(size_t)255; return q;
    };
    float* big0 = (float*)carve((size_t)NN * TT * 4);   // h ping
    float* big1 = (float*)carve((size_t)NN * TT * 4);   // m scratch
    float* big2 = (float*)carve((size_t)NN * TT * 4);   // agg scratch
    float* big3 = (float*)carve((size_t)NN * TT * 4);   // h pong / node_spatial
    float* scores = (float*)carve((size_t)NN * 4);
    int*   idxb   = (int*)carve((size_t)BB * TOPK * 4);
    float* gs     = (float*)carve((size_t)BB * TT * TOPK * 4);
    float* seq0   = (float*)carve((size_t)BB * TT * LSTM_H * 4);
    float* seq1   = (float*)carve((size_t)BB * TT * LSTM_H * 4);
    unsigned* cmax = (unsigned*)carve(64);

    float* q_out   = (float*)d_out;                     // [B,T,8]
    float* phi_out = (float*)d_out + BB * TT * MODE_N;  // [N,8]

    auto gemm = [&](int act, const float* A, const float* W,
                    const float* A2, const float* W2, const float* bias,
                    float* C, int M, int N, int K) {
        dim3 g((unsigned)((N + 63) / 64), (unsigned)(M / 128)), b(256);
        bool gn = (N % 64) != 0;
        if (act == 0) {
            if (gn) gemm_wmma<0, true ><<<g, b, 0, stream>>>(A, W, A2, W2, bias, C, M, N, K, K);
            else    gemm_wmma<0, false><<<g, b, 0, stream>>>(A, W, A2, W2, bias, C, M, N, K, K);
        } else if (act == 1) {
            if (gn) gemm_wmma<1, true ><<<g, b, 0, stream>>>(A, W, A2, W2, bias, C, M, N, K, K);
            else    gemm_wmma<1, false><<<g, b, 0, stream>>>(A, W, A2, W2, bias, C, M, N, K, K);
        } else {
            if (gn) gemm_wmma<2, true ><<<g, b, 0, stream>>>(A, W, A2, W2, bias, C, M, N, K, K);
            else    gemm_wmma<2, false><<<g, b, 0, stream>>>(A, W, A2, W2, bias, C, M, N, K, K);
        }
    };

    auto sage = [&](const float* X, int din, int dout,
                    const float* Wp, const float* bp, const float* Ws,
                    const float* Wn, const float* bb, float* out, int act) {
        // m = relu(X @ Wp + bp)
        gemm(1, X, Wp, nullptr, nullptr, bp, big1, NN, din, din);
        // agg = segment_max(m[src], dst)  (zero-fill == DGL empty fill)
        size_t aggN = (size_t)NN * din;
        zero_kernel<<<(unsigned)((aggN + 255) / 256), 256, 0, stream>>>(big2, aggN);
        long tot = (long)EE * (din >> 2);
        scatter_max<<<(unsigned)((tot + 255) / 256), 256, 0, stream>>>(
            big1, srcI, dstI, big2, din, EE);
        // out = act(X @ Ws + agg @ Wn + b)     (fused dual-GEMM)
        gemm(act, X, Ws, big2, Wn, bb, out, NN, dout, din);
    };

    // ---- GNN encoder
    sage(acc_Y, TT,   DIMD, c1_Wp, c1_bp, c1_Ws, c1_Wn, c1_b, big0, 2); // tanh
    sage(big0,  DIMD, DIMD, c2_Wp, c2_bp, c2_Ws, c2_Wn, c2_b, big3, 2); // tanh
    sage(big3,  DIMD, DIMD, c2_Wp, c2_bp, c2_Ws, c2_Wn, c2_b, big0, 2); // tanh
    sage(big0,  DIMD, TT,   c3_Wp, c3_bp, c3_Ws, c3_Wn, c3_b, big3, 0); // node_spatial
    // ---- per-node score + per-graph top-k
    sage(big3,  TT,   1,    sc_Wp, sc_bp, sc_Ws, sc_Wn, sc_b, scores, 0);
    topk_kernel<<<BB, 256, 0, stream>>>(scores, idxb);
    gather_sel<<<(BB * TT * TOPK + 255) / 256, 256, 0, stream>>>(big3, idxb, gs);

    // ---- LSTM stack (4 layers)
    lstm_layer<<<BB, 256, 0, stream>>>(gs,   l_Wih[0], l_Whh[0], l_bih[0], l_bhh[0], seq0, TOPK,   TT);
    lstm_layer<<<BB, 256, 0, stream>>>(seq0, l_Wih[1], l_Whh[1], l_bih[1], l_bhh[1], seq1, LSTM_H, TT);
    lstm_layer<<<BB, 256, 0, stream>>>(seq1, l_Wih[2], l_Whh[2], l_bih[2], l_bhh[2], seq0, LSTM_H, TT);
    lstm_layer<<<BB, 256, 0, stream>>>(seq0, l_Wih[3], l_Whh[3], l_bih[3], l_bhh[3], seq1, LSTM_H, TT);

    // ---- LSTM decoder MLP: [B*T,64] -> ... -> [B*T,8] = q
    const int MQ = BB * TT;
    gemm(2, seq1, de_W1, nullptr, nullptr, de_b1, big1, MQ, DIMD, LSTM_H);
    gemm(2, big1, de_W2, nullptr, nullptr, de_b2, big2, MQ, DIMD, DIMD);
    gemm(2, big2, de_W2, nullptr, nullptr, de_b2, big1, MQ, DIMD, DIMD);
    gemm(0, big1, de_W3, nullptr, nullptr, de_b3, q_out, MQ, MODE_N, DIMD);

    // ---- node decoder MLP: [N,256] -> ... -> [N,8] = phi, then col-norm
    gemm(2, big3, nd_W1, nullptr, nullptr, nd_b1, big1, NN, DIMD, TT);
    gemm(2, big1, nd_W2, nullptr, nullptr, nd_b2, big2, NN, DIMD, DIMD);
    gemm(2, big2, nd_W2, nullptr, nullptr, nd_b2, big1, NN, DIMD, DIMD);
    gemm(0, big1, nd_W3, nullptr, nullptr, nd_b3, phi_out, NN, MODE_N, DIMD);

    zero_kernel<<<1, 64, 0, stream>>>((float*)cmax, MODE_N);
    col_absmax<<<(NN * MODE_N + 255) / 256, 256, 0, stream>>>(phi_out, cmax, NN);
    col_scale <<<(NN * MODE_N + 255) / 256, 256, 0, stream>>>(phi_out, cmax, NN);
}
